// HyperGraphAttention_87136296501910
// MI455X (gfx1250) — matrix operand order repaired and checked by reading, not compiled, not run on hardware
//
#include <hip/hip_runtime.h>
#include <stdint.h>

#define DIM 128

typedef __attribute__((ext_vector_type(16))) __bf16 v16bf;
typedef __attribute__((ext_vector_type(8)))  float  v8f;

union BF16Frag { v16bf v; unsigned short s[16]; };

__device__ __forceinline__ unsigned short f32_to_bf16_rne(float f) {
  unsigned u = __float_as_uint(f);
  return (unsigned short)((u + 0x7FFFu + ((u >> 16) & 1u)) >> 16);
}
__device__ __forceinline__ float bf16_to_f32(unsigned short h) {
  return __uint_as_float(((unsigned)h) << 16);
}
__device__ __forceinline__ void split_bf16(float f, unsigned short &h, unsigned short &l) {
  h = f32_to_bf16_rne(f);
  float r = f - bf16_to_f32(h);
  l = f32_to_bf16_rne(r);
}

// ---------------------------------------------------------------------------
// Kernel 1: fold attention kernel into per-feature vectors p = K @ ka_slice
// ---------------------------------------------------------------------------
__global__ void hga_fold_att(const float* __restrict__ Ku, const float* __restrict__ Kv,
                             const float* __restrict__ Ke, const float* __restrict__ Ka,
                             float* __restrict__ pu, float* __restrict__ pv1,
                             float* __restrict__ pv2, float* __restrict__ pe) {
  int d = threadIdx.x;  // 128 threads
  float s0 = 0.f, s1 = 0.f, s2 = 0.f, s3 = 0.f;
  const float* ru = Ku + d * DIM;
  const float* rv = Kv + d * DIM;
  const float* re = Ke + d * DIM;
  for (int j = 0; j < DIM; ++j) {
    s0 = fmaf(ru[j], Ka[j],           s0);
    s1 = fmaf(rv[j], Ka[DIM + j],     s1);
    s2 = fmaf(rv[j], Ka[2 * DIM + j], s2);
    s3 = fmaf(re[j], Ka[3 * DIM + j], s3);
  }
  pu[d] = s0; pv1[d] = s1; pv2[d] = s2; pe[d] = s3;
}

// ---------------------------------------------------------------------------
// Kernel 2: per-row dot with one p vector (wave32: one wave per row)
// ---------------------------------------------------------------------------
__global__ void hga_rowdot1(const float* __restrict__ X, const float* __restrict__ p,
                            float* __restrict__ o, int n) {
  int lane = threadIdx.x & 31, wid = threadIdx.x >> 5;
  int row = blockIdx.x * 8 + wid;
  if (row >= n) return;
  float4 a = ((const float4*)(X + (size_t)row * DIM))[lane];
  float4 b = ((const float4*)p)[lane];
  float v = a.x * b.x + a.y * b.y + a.z * b.z + a.w * b.w;
#pragma unroll
  for (int off = 16; off > 0; off >>= 1) v += __shfl_xor(v, off);
  if (lane == 0) o[row] = v;
}

// Kernel 3: per-row dot with two p vectors (read video row once)
__global__ void hga_rowdot2(const float* __restrict__ X, const float* __restrict__ p1,
                            const float* __restrict__ p2, float* __restrict__ o1,
                            float* __restrict__ o2, int n) {
  int lane = threadIdx.x & 31, wid = threadIdx.x >> 5;
  int row = blockIdx.x * 8 + wid;
  if (row >= n) return;
  float4 a  = ((const float4*)(X + (size_t)row * DIM))[lane];
  float4 b1 = ((const float4*)p1)[lane];
  float4 b2 = ((const float4*)p2)[lane];
  float v1 = a.x * b1.x + a.y * b1.y + a.z * b1.z + a.w * b1.w;
  float v2 = a.x * b2.x + a.y * b2.y + a.z * b2.z + a.w * b2.w;
#pragma unroll
  for (int off = 16; off > 0; off >>= 1) { v1 += __shfl_xor(v1, off); v2 += __shfl_xor(v2, off); }
  if (lane == 0) { o1[row] = v1; o2[row] = v2; }
}

// ---------------------------------------------------------------------------
// Kernel 4: fused per-user attention + weighted edge aggregation.
// seg = hyper_edges[:,0] is sorted -> binary search this user's edge range.
// Reads each edge row exactly once; no atomics; deterministic.
// ---------------------------------------------------------------------------
__device__ __forceinline__ int seg_lower_bound(const long long* __restrict__ he, int E,
                                               long long key) {
  int lo = 0, hi = E;
  while (lo < hi) {
    int mid = (lo + hi) >> 1;
    if (he[3ll * mid] < key) lo = mid + 1; else hi = mid;
  }
  return lo;
}

__global__ __launch_bounds__(DIM) void hga_edge_attn_agg(
    const float* __restrict__ edges, const long long* __restrict__ he,
    const float* __restrict__ au, const float* __restrict__ av1,
    const float* __restrict__ av2, const float* __restrict__ pe,
    float* __restrict__ acc, int E) {
  int u = blockIdx.x, d = threadIdx.x;
  int lane = d & 31, wid = d >> 5;
  __shared__ float red[4];

  int lo = seg_lower_bound(he, E, (long long)u);
  int hi = seg_lower_bound(he, E, (long long)u + 1);

  float pd  = pe[d];
  float auu = au[u];
  float a = 0.f, ssum = 0.f;

  for (int e = lo; e < hi; ++e) {
    float x = edges[(size_t)e * DIM + d];        // coalesced 512B row
    float v = x * pd;                             // partial of ae = edge_row . p_edge
#pragma unroll
    for (int off = 16; off > 0; off >>= 1) v += __shfl_xor(v, off);
    if (lane == 0) red[wid] = v;
    __syncthreads();
    float ae = red[0] + red[1] + red[2] + red[3];
    __syncthreads();

    long long c1 = he[3ll * e + 1];
    long long c2 = he[3ll * e + 2];
    float sc = auu + av1[c1] + av2[c2] + ae;
    sc = sc > 0.f ? sc : 0.2f * sc;               // leaky_relu 0.2
    sc = fminf(2.f, fmaxf(-2.f, sc));             // clip
    float w = __expf(sc);
    ssum += w;                                    // same in all lanes
    a = fmaf(w, x, a);
  }
  acc[(size_t)u * DIM + d] = (hi > lo) ? (a / ssum) : 0.f;
}

// ---------------------------------------------------------------------------
// Kernel 5a: pre-swizzle kernel_edge (128x128, row-major K x N) into WMMA
// B-fragment order, split into bf16 hi/lo. Frag f = kt*8+nt; lane<16 holds
// column N=nt*16+lane, K = kt*32 + {0..7,16..23}; lane>=16 holds K = kt*32 + {8..15,24..31}.
// ---------------------------------------------------------------------------
__global__ void hga_pack_B(const float* __restrict__ Ke, unsigned short* __restrict__ Bh,
                           unsigned short* __restrict__ Bl) {
  int g = blockIdx.x * blockDim.x + threadIdx.x;  // 16384 total
  int e    = g & 15;
  int lane = (g >> 4) & 31;
  int f    = g >> 9;
  int kt = f >> 3, nt = f & 7;
  int k = kt * 32 + ((e >> 3) << 4) + ((lane >> 4) << 3) + (e & 7);
  int n = nt * 16 + (lane & 15);
  unsigned short h, l;
  split_bf16(Ke[k * DIM + n], h, l);
  Bh[g] = h; Bl[g] = l;
}

// ---------------------------------------------------------------------------
// Kernel 5: out = acc (nRows x 128) @ kernel_edge (128 x 128) via
// v_wmma_f32_16x16x32_bf16 with split-bf16 (A_hi*B_hi + A_hi*B_lo + A_lo*B_hi)
// for ~f32 accuracy. B fragments staged in LDS, shared by 8 waves/WG.
// One wave computes a 16-row x 128-col stripe: 8 ntiles x 4 ktiles x 3 = 96 WMMAs.
// ---------------------------------------------------------------------------
__global__ __launch_bounds__(256) void hga_out_gemm(
    const float* __restrict__ acc, const unsigned short* __restrict__ Bswz,
    float* __restrict__ out, int nRows, int nTiles) {
  __shared__ v16bf ldsB[2048];  // [0..1023] = hi frags, [1024..2047] = lo frags (64KB)
  {
    uint4* dst = (uint4*)ldsB;
    const uint4* src = (const uint4*)Bswz;
    for (int i = threadIdx.x; i < 4096; i += 256) dst[i] = src[i];
  }
  __syncthreads();

  int wave = threadIdx.x >> 5;
  int lane = threadIdx.x & 31;
  int tile = blockIdx.x * 8 + wave;
  if (tile >= nTiles) return;

  int M0   = tile * 16;
  int half = lane >> 4;       // K-half selector per ISA A layout
  int mr   = lane & 15;       // A row within tile
  int arow = M0 + mr; if (arow >= nRows) arow = nRows - 1;  // clamp (dup row, stores guarded)

  // Build A fragments: lanes 0-15 hold K {0..7,16..23}, lanes 16-31 hold K {8..15,24..31}
  const float* rp0 = acc + (size_t)arow * DIM + half * 8;
  v16bf ah[4], al[4];
#pragma unroll
  for (int kt = 0; kt < 4; ++kt) {
    const float* rp = rp0 + kt * 32;
    float4 x0 = *(const float4*)(rp);
    float4 x1 = *(const float4*)(rp + 4);
    float4 x2 = *(const float4*)(rp + 16);
    float4 x3 = *(const float4*)(rp + 20);
    float xs[16] = {x0.x, x0.y, x0.z, x0.w, x1.x, x1.y, x1.z, x1.w,
                    x2.x, x2.y, x2.z, x2.w, x3.x, x3.y, x3.z, x3.w};
    BF16Frag H, L;
#pragma unroll
    for (int i = 0; i < 16; ++i) split_bf16(xs[i], H.s[i], L.s[i]);
    ah[kt] = H.v; al[kt] = L.v;
  }

#pragma unroll
  for (int nt = 0; nt < 8; ++nt) {
    v8f c = {0.f, 0.f, 0.f, 0.f, 0.f, 0.f, 0.f, 0.f};
#pragma unroll
    for (int kt = 0; kt < 4; ++kt) {
      v16bf bh = ldsB[(kt * 8 + nt) * 32 + lane];
      v16bf bl = ldsB[1024 + (kt * 8 + nt) * 32 + lane];
      c = __builtin_amdgcn_wmma_f32_16x16x32_bf16(false, ah[kt], false, bh, (short)0, c, false, false);
      c = __builtin_amdgcn_wmma_f32_16x16x32_bf16(false, ah[kt], false, bl, (short)0, c, false, false);
      c = __builtin_amdgcn_wmma_f32_16x16x32_bf16(false, al[kt], false, bh, (short)0, c, false, false);
    }
    // C layout: VGPR v -> row M0+v (lanes 0-15) / M0+8+v (lanes 16-31), col = N tile + lane%16
    int col = nt * 16 + mr;
#pragma unroll
    for (int v = 0; v < 8; ++v) {
      int r = M0 + v + half * 8;
      if (r < nRows) out[(size_t)r * DIM + col] = c[v];
    }
  }
}

// ---------------------------------------------------------------------------
extern "C" void kernel_launch(void* const* d_in, const int* in_sizes, int n_in,
                              void* d_out, int out_size, void* d_ws, size_t ws_size,
                              hipStream_t stream) {
  (void)n_in; (void)out_size; (void)ws_size;
  const float*     users  = (const float*)d_in[0];
  const float*     videos = (const float*)d_in[1];
  const float*     edges  = (const float*)d_in[2];
  const long long* he     = (const long long*)d_in[3];
  const float*     Ku     = (const float*)d_in[4];
  const float*     Kv     = (const float*)d_in[5];
  const float*     Ke     = (const float*)d_in[6];
  const float*     Ka     = (const float*)d_in[7];

  const int nU = in_sizes[0] / DIM;
  const int nV = in_sizes[1] / DIM;
  const int E  = in_sizes[3] / 3;
  float* out = (float*)d_out;

  // Workspace carve-out (256B aligned slabs)
  uintptr_t w = (uintptr_t)d_ws;
  auto alignup = [](uintptr_t x) { return (x + 255) & ~(uintptr_t)255; };
  float* pu  = (float*)w;   w += (size_t)4 * DIM * sizeof(float);   // pu,pv1,pv2,pe
  float* pv1 = pu + DIM;
  float* pv2 = pu + 2 * DIM;
  float* pe  = pu + 3 * DIM;
  w = alignup(w);
  float* au  = (float*)w;   w += (size_t)nU * sizeof(float);  w = alignup(w);
  float* av1 = (float*)w;   w += (size_t)nV * sizeof(float);  w = alignup(w);
  float* av2 = (float*)w;   w += (size_t)nV * sizeof(float);  w = alignup(w);
  float* acc = (float*)w;   w += (size_t)nU * DIM * sizeof(float); w = alignup(w);
  unsigned short* Bh = (unsigned short*)w;   // hi frags (32KB) immediately followed by
  unsigned short* Bl = Bh + 16384;           // lo frags (32KB) -> one contiguous LDS copy

  hga_fold_att<<<1, DIM, 0, stream>>>(Ku, Kv, Ke, Ka, pu, pv1, pv2, pe);
  hga_rowdot1<<<(nU + 7) / 8, 256, 0, stream>>>(users, pu, au, nU);
  hga_rowdot2<<<(nV + 7) / 8, 256, 0, stream>>>(videos, pv1, pv2, av1, av2, nV);
  hga_pack_B<<<64, 256, 0, stream>>>(Ke, Bh, Bl);
  hga_edge_attn_agg<<<nU, DIM, 0, stream>>>(edges, he, au, av1, av2, pe, acc, E);

  int nTiles = (nU + 15) / 16;
  hga_out_gemm<<<(nTiles + 7) / 8, 256, 0, stream>>>(acc, Bh, out, nU, nTiles);
}
